// SimpleSelfAttention_35862976921882
// MI455X (gfx1250) — matrix operand order repaired and checked by reading, not compiled, not run on hardware
//
#include <hip/hip_runtime.h>
#include <hip/hip_bf16.h>

#define D_EMBED 256
#define NROW    16384   // B * S
#define SEQ     4096

typedef __bf16 bf16_t;
typedef __bf16 v16bf __attribute__((ext_vector_type(16)));
typedef float  v8f   __attribute__((ext_vector_type(8)));
typedef unsigned int v4u __attribute__((ext_vector_type(4)));
typedef int          v4i __attribute__((ext_vector_type(4)));

// ---- CDNA5 data-movement feature detection (device pass only) --------------
#define HAS_ASYNC_LDS __has_builtin(__builtin_amdgcn_global_load_async_to_lds_b128)
#define HAS_TDM       __has_builtin(__builtin_amdgcn_tensor_load_to_lds)

#if defined(__gfx1250__)
#if HAS_TDM
#pragma message("CDNA5 feature report: tensor_load_to_lds AVAILABLE (TDM path in use)")
#else
#pragma message("CDNA5 feature report: tensor_load_to_lds NOT available")
#endif
#if HAS_ASYNC_LDS
#pragma message("CDNA5 feature report: global_load_async_to_lds_b128 AVAILABLE")
#else
#pragma message("CDNA5 feature report: global_load_async_to_lds_b128 NOT available")
#endif
#endif

#if HAS_TDM
#if __has_include(<hip/amd_detail/amd_gfx1250_TDM.h>)
#define TDM_6ARG 1
#if defined(__gfx1250__)
#pragma message("CDNA5 feature report: TDM builtin uses 6-arg form")
#endif
#else
#define TDM_6ARG 0
#if defined(__gfx1250__)
#pragma message("CDNA5 feature report: TDM builtin uses 5-arg form")
#endif
#endif
typedef unsigned int u32x4 __attribute__((ext_vector_type(4)));
typedef int          i32x4 __attribute__((ext_vector_type(4)));
typedef int          i32x8 __attribute__((ext_vector_type(8)));

// 2-D tile load via Tensor Data Mover. D# built per CDNA5 ISA.
// data_size = 2 bytes (bf16). Issued by one wave; tracked with TENSORcnt.
__device__ __forceinline__ void tdm_load_2d(unsigned lds_off, unsigned long long gaddr,
                                            unsigned tile_d0, unsigned tile_d1,
                                            unsigned tensor_d0, unsigned tensor_d1,
                                            unsigned stride_elems) {
  u32x4 g0;
  g0[0] = 1u;                                            // count=1, user mode
  g0[1] = lds_off;                                       // lds_addr (bytes)
  g0[2] = (unsigned)(gaddr & 0xffffffffull);             // global_addr[31:0]
  g0[3] = (unsigned)((gaddr >> 32) & 0x01ffffffull) | (2u << 30);  // addr[56:32] | type=2
  i32x8 g1;
  g1[0] = (int)(1u << 16);                               // data_size=1 (2 bytes)
  g1[1] = (int)((tensor_d0 & 0xffffu) << 16);            // tensor_dim0[15:0]
  g1[2] = (int)(((tensor_d0 >> 16) & 0xffffu) | ((tensor_d1 & 0xffffu) << 16));
  g1[3] = (int)(((tensor_d1 >> 16) & 0xffffu) | ((tile_d0 & 0xffffu) << 16));
  g1[4] = (int)(tile_d1 & 0xffffu);                      // tile_dim1, tile_dim2=0
  g1[5] = (int)stride_elems;                             // tensor_dim0_stride[31:0]
  g1[6] = 0;
  g1[7] = 0;
  i32x4 z4 = {0, 0, 0, 0};
#if TDM_6ARG
  i32x8 z8 = {0, 0, 0, 0, 0, 0, 0, 0};
  __builtin_amdgcn_tensor_load_to_lds(g0, g1, z4, z4, z8, 0);
#else
  __builtin_amdgcn_tensor_load_to_lds(g0, g1, z4, z4, 0);
#endif
}
#endif  // HAS_TDM

// 16-byte global->LDS copy: async (ASYNCcnt) when available, else via regs.
// Builtin signature (probe-confirmed by diagnostic): (v4i AS1*, v4i AS3*, Ii, Ii)
__device__ __forceinline__ void copy16_to_lds(const bf16_t* g, bf16_t* l) {
#if HAS_ASYNC_LDS
  __builtin_amdgcn_global_load_async_to_lds_b128(
      (__attribute__((address_space(1))) v4i*)(unsigned long long)(size_t)g,
      (__attribute__((address_space(3))) v4i*)(unsigned)(size_t)l, 0, 0);
#else
  *(v4u*)l = *(const v4u*)g;
#endif
}
__device__ __forceinline__ void copy_wait() {
#if HAS_ASYNC_LDS
  asm volatile("s_wait_asynccnt 0x0" ::: "memory");
#endif
}
__device__ __forceinline__ unsigned lds_off_u32(const void* p) {
  return (unsigned)(size_t)p;  // low 32 bits of flat shared address = LDS offset
}

// One WMMA operand fragment: 16 bf16 per lane = 8 VGPRs, loadable as 2x16B.
union Frag { v16bf v; v4u q[2]; };

__device__ __forceinline__ v8f wmma_bf16(const Frag& a, const Frag& b, v8f c) {
  return __builtin_amdgcn_wmma_f32_16x16x32_bf16(false, a.v, false, b.v,
                                                 (short)0, c, false, false);
}

// ---------------------------------------------------------------- cast f32->bf16
__global__ __launch_bounds__(256) void k_cast_bf16(const float* __restrict__ src,
                                                   bf16_t* __restrict__ dst, int n) {
  int base = (blockIdx.x * blockDim.x + threadIdx.x) * 4;
  if (base < n) {
#pragma unroll
    for (int j = 0; j < 4; ++j) dst[base + j] = (bf16_t)src[base + j];
  }
}

// ---------------------------------------------------------------- GEMM: out = A * W^T + bias
// W tile (64 out-cols x 256) staged in LDS once per block (TDM / async / manual),
// K-loop double-buffered in registers so loads overlap WMMA.
// mode 0: bf16 out; mode 1: bf16 out + relu; mode 2: f32 out
__global__ __launch_bounds__(256) void k_gemm(const bf16_t* __restrict__ A,
                                              const bf16_t* __restrict__ W,
                                              const float* __restrict__ bias,
                                              bf16_t* __restrict__ outB,
                                              float* __restrict__ outF, int mode) {
  __shared__ bf16_t ldsW[64 * D_EMBED];  // 32 KB
  const int lane = threadIdx.x & 31;
  const int wave = threadIdx.x >> 5;
  const int half = lane >> 4;
  const int l16  = lane & 15;
  const int row0 = blockIdx.x * 128 + wave * 16;
  const int col0 = blockIdx.y * 64;

#if HAS_TDM
  if (wave == 0) {
    tdm_load_2d(lds_off_u32(ldsW),
                (unsigned long long)(size_t)(W + (size_t)col0 * D_EMBED),
                D_EMBED, 64, D_EMBED, D_EMBED, D_EMBED);
    asm volatile("s_wait_tensorcnt 0x0" ::: "memory");
  }
#else
  {
    const int r  = threadIdx.x >> 2;        // 0..63
    const int c0 = (threadIdx.x & 3) * 64;  // 0..192
    const bf16_t* src = W + (size_t)(col0 + r) * D_EMBED + c0;
    bf16_t* dst = ldsW + r * D_EMBED + c0;
#pragma unroll
    for (int i = 0; i < 8; ++i) copy16_to_lds(src + i * 8, dst + i * 8);
  }
  copy_wait();
#endif
  __syncthreads();

  const bf16_t* arow = A + (size_t)(row0 + l16) * D_EMBED;
  Frag a[2];
  Frag b[2][4];
#define LOAD_A(buf, kk)                                                  \
  do {                                                                   \
    a[buf].q[0] = *(const v4u*)(arow + (kk) + half * 16);                \
    a[buf].q[1] = *(const v4u*)(arow + (kk) + half * 16 + 8);            \
  } while (0)
#define LOAD_B(buf, kk)                                                  \
  do {                                                                   \
    _Pragma("unroll") for (int t = 0; t < 4; ++t) {                      \
      const bf16_t* wr = ldsW + (t * 16 + l16) * D_EMBED + (kk) + half * 16; \
      b[buf][t].q[0] = ((const v4u*)wr)[0];                              \
      b[buf][t].q[1] = ((const v4u*)wr)[1];                              \
    }                                                                    \
  } while (0)

  LOAD_A(0, 0);
  LOAD_B(0, 0);
  v8f acc[4] = {};
#pragma unroll
  for (int s = 0; s < 8; ++s) {
    const int cur = s & 1;
    if (s < 7) {               // prefetch next K-step while WMMAs run
      LOAD_A(cur ^ 1, (s + 1) * 32);
      LOAD_B(cur ^ 1, (s + 1) * 32);
    }
#pragma unroll
    for (int t = 0; t < 4; ++t) acc[t] = wmma_bf16(a[cur], b[cur][t], acc[t]);
  }
#undef LOAD_A
#undef LOAD_B

#pragma unroll
  for (int t = 0; t < 4; ++t) {
    int col = col0 + t * 16 + l16;
    float bv = bias[col];
#pragma unroll
    for (int v = 0; v < 8; ++v) {
      int row = row0 + v + 8 * half;
      float r = acc[t][v] + bv;
      if (mode == 1) r = fmaxf(r, 0.0f);
      if (mode == 2) outF[(size_t)row * D_EMBED + col] = r;
      else           outB[(size_t)row * D_EMBED + col] = (bf16_t)r;
    }
  }
}

// ---------------------------------------------------------------- flash attention
// Per block: 8 waves x 16 query rows = 128 rows of one batch; 32 kv rows/step.
__global__ __launch_bounds__(256) void k_flash_attn(const bf16_t* __restrict__ Q,
                                                    const bf16_t* __restrict__ Km,
                                                    const bf16_t* __restrict__ Vm,
                                                    float* __restrict__ O) {
  __shared__ bf16_t ldsK[32 * D_EMBED];    // kv rows, row-major        (16 KB)
  __shared__ bf16_t ldsVt[D_EMBED * 32];   // V transposed: [embed][kv] (16 KB)
  __shared__ bf16_t ldsP[8][16 * 32];      // per-wave P staging        ( 8 KB)

  const int lane = threadIdx.x & 31;
  const int wave = threadIdx.x >> 5;
  const int half = lane >> 4;
  const int l16  = lane & 15;
  const int batch = blockIdx.x >> 5;
  const int qrow0 = batch * SEQ + (blockIdx.x & 31) * 128 + wave * 16;

  Frag qf[8];  // whole 16x256 Q row-block in registers
  const bf16_t* qrow = Q + (size_t)(qrow0 + l16) * D_EMBED;
#pragma unroll
  for (int kk = 0; kk < 8; ++kk) {
    qf[kk].q[0] = *(const v4u*)(qrow + kk * 32 + half * 16);
    qf[kk].q[1] = *(const v4u*)(qrow + kk * 32 + half * 16 + 8);
  }

  v8f oacc[16] = {};
  float mrow[8], lrow[8];
#pragma unroll
  for (int v = 0; v < 8; ++v) { mrow[v] = -__builtin_inff(); lrow[v] = 0.0f; }

  const bf16_t* Kb = Km + (size_t)batch * SEQ * D_EMBED;
  const bf16_t* Vb = Vm + (size_t)batch * SEQ * D_EMBED;

  for (int kt = 0; kt < SEQ / 32; ++kt) {
    __syncthreads();
    if (kt + 2 < SEQ / 32) {  // global_prefetch_b8 two tiles ahead into GL2
      size_t nxt = (size_t)(kt + 2) * 32 * D_EMBED + (size_t)threadIdx.x * 32;
      __builtin_prefetch(Kb + nxt, 0, 1);
      __builtin_prefetch(Vb + nxt, 0, 1);
    }
#if HAS_TDM
    if (wave == 0) {  // one TDM op moves the whole 32x256 K tile
      tdm_load_2d(lds_off_u32(ldsK),
                  (unsigned long long)(size_t)(Kb + (size_t)kt * 32 * D_EMBED),
                  D_EMBED, 32, D_EMBED, SEQ, D_EMBED);
      asm volatile("s_wait_tensorcnt 0x0" ::: "memory");
    }
#else
    {
      const int r  = threadIdx.x >> 3;
      const int c0 = (threadIdx.x & 7) * 32;
      const bf16_t* ksrc = Kb + (size_t)(kt * 32 + r) * D_EMBED + c0;
      bf16_t* kdst = ldsK + r * D_EMBED + c0;
#pragma unroll
      for (int i = 0; i < 4; ++i) copy16_to_lds(ksrc + i * 8, kdst + i * 8);
    }
#endif
    {  // V transpose: 2 kv rows per thread, packed dword LDS stores
      const int r2 = (threadIdx.x >> 4) * 2;   // 0,2,..,30
      const int c0 = (threadIdx.x & 15) * 16;  // 0..240
      union { v4u q[2]; bf16_t h[16]; } va, vb2;
      const bf16_t* v0 = Vb + (size_t)(kt * 32 + r2) * D_EMBED + c0;
      va.q[0]  = ((const v4u*)v0)[0];
      va.q[1]  = ((const v4u*)v0)[1];
      vb2.q[0] = ((const v4u*)(v0 + D_EMBED))[0];
      vb2.q[1] = ((const v4u*)(v0 + D_EMBED))[1];
#pragma unroll
      for (int i = 0; i < 16; ++i) {
        union { bf16_t h[2]; unsigned u; } p;
        p.h[0] = va.h[i];
        p.h[1] = vb2.h[i];
        *(unsigned*)(ldsVt + (c0 + i) * 32 + r2) = p.u;
      }
    }
#if !HAS_TDM
    copy_wait();
#endif
    __syncthreads();

    // S = Q * K^T : two 16x16 f32 tiles (16 WMMA)
    v8f sc[2] = {};
#pragma unroll
    for (int t = 0; t < 2; ++t)
#pragma unroll
      for (int kk = 0; kk < 8; ++kk) {
        Frag b;
        const bf16_t* kr = ldsK + (t * 16 + l16) * D_EMBED + kk * 32 + half * 16;
        b.q[0] = ((const v4u*)kr)[0];
        b.q[1] = ((const v4u*)kr)[1];
        sc[t] = wmma_bf16(qf[kk], b, sc[t]);
      }

    // online softmax (row = v + 8*half)
    float corr[8];
    bf16_t* pb = ldsP[wave];
#pragma unroll
    for (int v = 0; v < 8; ++v) {
      float a0 = sc[0][v] * 0.0625f;  // 1/sqrt(256)
      float a1 = sc[1][v] * 0.0625f;
      float mx = fmaxf(a0, a1);
#pragma unroll
      for (int off = 1; off < 16; off <<= 1) mx = fmaxf(mx, __shfl_xor(mx, off, 32));
      float mnew = fmaxf(mrow[v], mx);
      float c  = __expf(mrow[v] - mnew);
      float p0 = __expf(a0 - mnew);
      float p1 = __expf(a1 - mnew);
      float rs = p0 + p1;
#pragma unroll
      for (int off = 1; off < 16; off <<= 1) rs += __shfl_xor(rs, off, 32);
      lrow[v] = lrow[v] * c + rs;
      mrow[v] = mnew;
      corr[v] = c;
      pb[(v + 8 * half) * 32 + l16]      = (bf16_t)p0;
      pb[(v + 8 * half) * 32 + 16 + l16] = (bf16_t)p1;
    }
#pragma unroll
    for (int t = 0; t < 16; ++t)
#pragma unroll
      for (int v = 0; v < 8; ++v) oacc[t][v] *= corr[v];

    asm volatile("s_wait_dscnt 0x0" ::: "memory");  // wave-synchronous P transpose

    Frag pa;  // P as A-matrix 16x32
    const bf16_t* pr = pb + l16 * 32;
    pa.q[0] = *(const v4u*)(pr + half * 16);
    pa.q[1] = *(const v4u*)(pr + half * 16 + 8);
#pragma unroll
    for (int t = 0; t < 16; ++t) {
      Frag b;  // column n of V = Vt row (t*16 + l16), contiguous 32 B
      const bf16_t* vr = ldsVt + (t * 16 + l16) * 32 + half * 16;
      b.q[0] = ((const v4u*)vr)[0];
      b.q[1] = ((const v4u*)vr)[1];
      oacc[t] = wmma_bf16(pa, b, oacc[t]);
    }
  }

#pragma unroll
  for (int t = 0; t < 16; ++t)
#pragma unroll
    for (int v = 0; v < 8; ++v) {
      int row = qrow0 + v + 8 * half;
      int col = t * 16 + l16;
      O[(size_t)row * D_EMBED + col] = oacc[t][v] / lrow[v];
    }
}

// ---------------------------------------------------------------- residual + layernorm
__global__ __launch_bounds__(256) void k_add_ln(const float* __restrict__ X,
                                                const float* __restrict__ Y,
                                                const float* __restrict__ g,
                                                const float* __restrict__ be,
                                                bf16_t* __restrict__ outB,
                                                float* __restrict__ outF, int mode) {
  const int lane = threadIdx.x & 31;
  const int wave = threadIdx.x >> 5;
  const int row  = blockIdx.x * 8 + wave;
  const float* xr = X + (size_t)row * D_EMBED;
  const float* yr = Y + (size_t)row * D_EMBED;
  float vals[8];
  float s = 0.0f;
#pragma unroll
  for (int j = 0; j < 8; ++j) { int c = lane + j * 32; vals[j] = xr[c] + yr[c]; s += vals[j]; }
#pragma unroll
  for (int off = 1; off < 32; off <<= 1) s += __shfl_xor(s, off, 32);
  float mu = s * (1.0f / 256.0f);
  float vv = 0.0f;
#pragma unroll
  for (int j = 0; j < 8; ++j) { float d = vals[j] - mu; vv += d * d; }
#pragma unroll
  for (int off = 1; off < 32; off <<= 1) vv += __shfl_xor(vv, off, 32);
  float rinv = rsqrtf(vv * (1.0f / 256.0f) + 1e-5f);
#pragma unroll
  for (int j = 0; j < 8; ++j) {
    int c = lane + j * 32;
    float r = (vals[j] - mu) * rinv * g[c] + be[c];
    if (mode == 0) outB[(size_t)row * D_EMBED + c] = (bf16_t)r;
    else           outF[(size_t)row * D_EMBED + c] = r;
  }
}

// ---------------------------------------------------------------- launch
extern "C" void kernel_launch(void* const* d_in, const int* in_sizes, int n_in,
                              void* d_out, int out_size, void* d_ws, size_t ws_size,
                              hipStream_t stream) {
  (void)in_sizes; (void)n_in; (void)out_size; (void)ws_size;
  const float* x   = (const float*)d_in[0];
  const float* Wq  = (const float*)d_in[1];
  const float* bq  = (const float*)d_in[2];
  const float* Wk  = (const float*)d_in[3];
  const float* bk  = (const float*)d_in[4];
  const float* Wv  = (const float*)d_in[5];
  const float* bv  = (const float*)d_in[6];
  const float* Wl  = (const float*)d_in[7];
  const float* bl  = (const float*)d_in[8];
  const float* g1  = (const float*)d_in[9];
  const float* be1 = (const float*)d_in[10];
  const float* g2  = (const float*)d_in[11];
  const float* be2 = (const float*)d_in[12];
  float* out = (float*)d_out;

  char* ws = (char*)d_ws;
  const size_t SZB = (size_t)NROW * D_EMBED * sizeof(bf16_t);      // 8 MiB
  const size_t SZF = (size_t)NROW * D_EMBED * sizeof(float);       // 16 MiB
  const size_t WSZ = (size_t)D_EMBED * D_EMBED * sizeof(bf16_t);   // 128 KiB
  bf16_t* xb  = (bf16_t*)(ws);
  bf16_t* qb  = (bf16_t*)(ws + 1 * SZB);
  bf16_t* kb  = (bf16_t*)(ws + 2 * SZB);
  bf16_t* vb  = (bf16_t*)(ws + 3 * SZB);
  float*  aof = (float*) (ws + 4 * SZB);             // attn out (f32)
  bf16_t* h1b = (bf16_t*)(ws + 4 * SZB + SZF);
  bf16_t* hmb = (bf16_t*)(ws + 5 * SZB + SZF);
  float*  h2f = aof;                                  // reuse: aof consumed by LN1
  bf16_t* wqb = (bf16_t*)(ws + 6 * SZB + SZF);
  bf16_t* wkb = (bf16_t*)(ws + 6 * SZB + SZF + 1 * WSZ);
  bf16_t* wvb = (bf16_t*)(ws + 6 * SZB + SZF + 2 * WSZ);
  bf16_t* wlb = (bf16_t*)(ws + 6 * SZB + SZF + 3 * WSZ);

  const int nx = NROW * D_EMBED;
  const int nw = D_EMBED * D_EMBED;
  k_cast_bf16<<<dim3((nx / 4 + 255) / 256), dim3(256), 0, stream>>>(x,  xb,  nx);
  k_cast_bf16<<<dim3((nw / 4 + 255) / 256), dim3(256), 0, stream>>>(Wq, wqb, nw);
  k_cast_bf16<<<dim3((nw / 4 + 255) / 256), dim3(256), 0, stream>>>(Wk, wkb, nw);
  k_cast_bf16<<<dim3((nw / 4 + 255) / 256), dim3(256), 0, stream>>>(Wv, wvb, nw);
  k_cast_bf16<<<dim3((nw / 4 + 255) / 256), dim3(256), 0, stream>>>(Wl, wlb, nw);

  dim3 gg(NROW / 128, D_EMBED / 64);
  k_gemm<<<gg, dim3(256), 0, stream>>>(xb, wqb, bq, qb, nullptr, 0);
  k_gemm<<<gg, dim3(256), 0, stream>>>(xb, wkb, bk, kb, nullptr, 0);
  k_gemm<<<gg, dim3(256), 0, stream>>>(xb, wvb, bv, vb, nullptr, 0);

  k_flash_attn<<<dim3(NROW / 128), dim3(256), 0, stream>>>(qb, kb, vb, aof);

  k_add_ln<<<dim3(NROW / 8), dim3(256), 0, stream>>>(x, aof, g1, be1, h1b, nullptr, 0);
  k_gemm<<<gg, dim3(256), 0, stream>>>(h1b, wlb, bl, hmb, nullptr, 1);
  k_gemm<<<gg, dim3(256), 0, stream>>>(hmb, wlb, bl, nullptr, h2f, 2);
  k_add_ln<<<dim3(NROW / 8), dim3(256), 0, stream>>>(x, h2f, g2, be2, nullptr, out, 1);
}
